// SAWB_2740189134896
// MI455X (gfx1250) — 8-bit matrix operand order repaired and checked by reading, compiled, not run on hardware
//
#include <hip/hip_runtime.h>
#include <stdint.h>

typedef int v8i __attribute__((ext_vector_type(8)));

#define BATCH  8192
#define LOUT   92
#define MTOT   (BATCH*LOUT)
#define BNEPS  1e-5f

struct Layer { int Cin, Cout, ksz, pad, Kreal, KC, nblk, fragOff, swOff; };

__constant__ Layer LY[7] = {
  {24, 24,11,5,264,5,2,     0,   0},
  {24, 24,11,5,264,5,2,  2560,  24},
  {24,128, 5,2,120,2,8,  5120,  48},
  {128,24, 3,1,384,6,2,  9216, 176},
  {24, 24, 3,1, 72,2,2, 12288, 200},
  {24, 24, 3,1, 72,2,2, 13312, 224},
  {24, 24, 3,1, 72,2,2, 14336, 248},
};
static const Layer HLY[7] = {
  {24, 24,11,5,264,5,2,     0,   0},
  {24, 24,11,5,264,5,2,  2560,  24},
  {24,128, 5,2,120,2,8,  5120,  48},
  {128,24, 3,1,384,6,2,  9216, 176},
  {24, 24, 3,1, 72,2,2, 12288, 200},
  {24, 24, 3,1, 72,2,2, 13312, 224},
  {24, 24, 3,1, 72,2,2, 14336, 248},
};

struct Ptrs7 { const float* p[7]; };

__device__ __forceinline__ float redu64(float v, float* red, int tid) {
  red[tid] = v; __syncthreads();
  #pragma unroll
  for (int o = 32; o > 0; o >>= 1) { if (tid < o) red[tid] += red[tid + o]; __syncthreads(); }
  float r = red[0]; __syncthreads();
  return r;
}

__global__ void k_prep(Ptrs7 wp, int* __restrict__ qfrag, float* __restrict__ sw) {
  __shared__ float red[64];
  int tid = threadIdx.x;
  int blk = blockIdx.x;
  int layer = 0, c = blk;
  while (c >= LY[layer].Cout) { c -= LY[layer].Cout; layer++; }
  Layer L = LY[layer];
  const float* w = wp.p[layer] + (size_t)c * L.Cin * L.ksz;
  int n = L.Cin * L.ksz;

  float s1 = 0.f, s2 = 0.f;
  for (int i = tid; i < n; i += 64) { float v = w[i]; s1 += v; s2 += v * v; }
  float sum   = redu64(s1, red, tid);
  float sumsq = redu64(s2, red, tid);
  float m   = sum / (float)n;
  float var = (sumsq - sum * sum / (float)n) / (float)(n - 1);
  float sd  = sqrtf(var);

  float sa = 0.f;
  for (int i = tid; i < n; i += 64) sa += fabsf(w[i] - m);
  float asum = redu64(sa, red, tid);
  float meanabs = asum / ((float)n * sd);
  float swv = exp2f(rintf(log2f(meanabs)));
  if (tid == 0) sw[L.swOff + c] = swv;

  int nb = c >> 4, nl = c & 15;
  int items = L.KC * 16;
  for (int i = tid; i < items; i += 64) {
    int kc = i >> 4; int r = i & 15; int half = r >> 3; int v = r & 7;
    unsigned d = 0;
    for (int j = 0; j < 4; j++) {
      int kk = kc * 64 + ((v >= 4) ? 32 : 0) + half * 16 + (v & 3) * 4 + j;
      int bval = 0;
      if (kk < L.Kreal) {
        int t = kk / L.Cin, ci = kk - t * L.Cin;
        float bw = w[ci * L.ksz + t] - m;
        bval = (bw > 0.f) - (bw < 0.f);
      }
      d |= ((unsigned)(bval & 0xFF)) << (8 * j);
    }
    qfrag[(size_t)L.fragOff + (((size_t)(nb * L.KC + kc) * 32) + (half * 16 + nl)) * 8 + v] = (int)d;
  }
}

__global__ void k_conv1(const float* __restrict__ x, const float* __restrict__ w1,
                        const float* __restrict__ b1, float* __restrict__ z) {
  __shared__ float wsm[24 * 11];
  __shared__ float bsm[24];
  int tid = threadIdx.x;
  for (int i = tid; i < 264; i += 256) wsm[i] = w1[i];
  if (tid < 24) bsm[tid] = b1[tid];
  __syncthreads();
  int mr = blockIdx.x * 256 + tid;
  if (mr >= MTOT) return;
  int b = mr / LOUT, p = mr % LOUT;
  const float* xr = x + (size_t)b * 102 + p;
  float acc[24];
  #pragma unroll
  for (int c = 0; c < 24; c++) acc[c] = bsm[c];
  for (int t = 0; t < 11; t++) {
    float xv = xr[t];
    #pragma unroll
    for (int c = 0; c < 24; c++) acc[c] += xv * wsm[c * 11 + t];
  }
  float* zr = z + (size_t)b * 24 * LOUT + p;
  #pragma unroll
  for (int c = 0; c < 24; c++) zr[c * LOUT] = acc[c];
}

__global__ void k_stats(const float* __restrict__ z, int C, float* __restrict__ out) {
  __shared__ double red[256];
  int c = blockIdx.x, tid = threadIdx.x;
  double s1 = 0.0, s2 = 0.0;
  for (int b = tid; b < BATCH; b += 256) {
    const float* row = z + ((size_t)b * C + c) * LOUT;
    for (int p = 0; p < LOUT; p++) { double v = row[p]; s1 += v; s2 += v * v; }
  }
  red[tid] = s1; __syncthreads();
  for (int o = 128; o > 0; o >>= 1) { if (tid < o) red[tid] += red[tid + o]; __syncthreads(); }
  double S1 = red[0]; __syncthreads();
  red[tid] = s2; __syncthreads();
  for (int o = 128; o > 0; o >>= 1) { if (tid < o) red[tid] += red[tid + o]; __syncthreads(); }
  double S2 = red[0];
  if (tid == 0) {
    double N = (double)MTOT;
    double m = S1 / N;
    double var = S2 / N - m * m;
    out[c]       = (float)m;
    out[128 + c] = rsqrtf((float)var + BNEPS);
  }
}

__global__ void k_bin(const float* __restrict__ z, const float* __restrict__ st,
                      int8_t* __restrict__ a, int C, int total) {
  int i = blockIdx.x * 256 + threadIdx.x;
  if (i >= total) return;
  int c  = i % C;
  int bp = i / C;
  int b  = bp / LOUT, p = bp % LOUT;
  float d = z[((size_t)b * C + c) * LOUT + p] - st[c];
  a[i] = (int8_t)((d > 0.f) - (d < 0.f));
}

__global__ void k_res(const float* __restrict__ za, const float* __restrict__ sa,
                      const float* __restrict__ zb, const float* __restrict__ sb,
                      float* __restrict__ o, int total) {
  int i = blockIdx.x * 256 + threadIdx.x;
  if (i >= total) return;
  int c = (i / LOUT) % 24;
  o[i] = (za[i] - sa[c]) * sa[128 + c] + (zb[i] - sb[c]) * sb[128 + c];
}

template <int CP, int KC>
__global__ void k_binconv(const int8_t* __restrict__ act, const int* __restrict__ qfrag,
                          const float* __restrict__ sw, float* __restrict__ z,
                          int Cout, int pad) {
  constexpr int TMAX   = (KC * 64 - 1) / CP;
  constexpr int W      = LOUT + TMAX;
  constexpr int ROWDW  = W * CP / 4;
  constexpr int DATADW = LOUT * CP / 4;
  __shared__ int ldsA[4 * ROWDW];
  __shared__ int ldsB[KC * 256];

  int tid = threadIdx.x;
  int blockM0 = blockIdx.x * 256;
  int b0 = blockM0 / LOUT;
  int nb = (blockM0 + 255) / LOUT - b0 + 1;
  int padDW = pad * CP / 4;

  const int* actI = (const int*)act;
  for (int d = tid; d < nb * ROWDW; d += 256) {
    int r = d / ROWDW, w = d - r * ROWDW;
    int val = 0;
    int wd = w - padDW;
    if (wd >= 0 && wd < DATADW) val = actI[(size_t)(b0 + r) * DATADW + wd];
    ldsA[d] = val;
  }
  const int* qb = qfrag + (size_t)blockIdx.y * KC * 256;
  for (int d = tid; d < KC * 256; d += 256) ldsB[d] = qb[d];
  __syncthreads();

  int wave = tid >> 5, lane = tid & 31;
  int half = lane >> 4, ml = lane & 15;
  int mbase = blockM0 + wave * 32;
  int mrow0 = mbase + ml, mrow1 = mbase + 16 + ml;
  int bA = mrow0 / LOUT, pA = mrow0 % LOUT;
  int bB = mrow1 / LOUT, pB = mrow1 % LOUT;
  int rowBaseA = (bA - b0) * ROWDW;
  int rowBaseB = (bB - b0) * ROWDW;

  v8i acc0, acc1;
  #pragma unroll
  for (int i = 0; i < 8; i++) { acc0[i] = 0; acc1[i] = 0; }

  #pragma unroll
  for (int kc = 0; kc < KC; kc++) {
    v8i bf, af0, af1;
    #pragma unroll
    for (int v = 0; v < 8; v++) bf[v] = ldsB[kc * 256 + lane * 8 + v];
    #pragma unroll
    for (int v = 0; v < 8; v++) {
      int kb = kc * 64 + ((v & 4) << 3) + ((v & 2) << 2) + ((v & 1) << 2) + (half << 4);
      int t = kb / CP, ci = kb - t * CP;
      af0[v] = ldsA[rowBaseA + (((pA + t) * CP + ci) >> 2)];
      af1[v] = ldsA[rowBaseB + (((pB + t) * CP + ci) >> 2)];
    }
    acc0 = __builtin_amdgcn_wmma_i32_16x16x64_iu8(true, af0, true, bf, acc0, false, false);
    acc1 = __builtin_amdgcn_wmma_i32_16x16x64_iu8(true, af1, true, bf, acc1, false, false);
  }

  int c = blockIdx.y * 16 + ml;
  if (c < Cout) {
    float s = sw[c];
    #pragma unroll
    for (int r = 0; r < 8; r++) {
      int mo = mbase + r + half * 8;
      int bo = mo / LOUT, po = mo % LOUT;
      z[((size_t)bo * Cout + c) * LOUT + po] = s * (float)acc0[r];
      int mo1 = mo + 16;
      int bo1 = mo1 / LOUT, po1 = mo1 % LOUT;
      z[((size_t)bo1 * Cout + c) * LOUT + po1] = s * (float)acc1[r];
    }
  }
}

__global__ void k_fc(const float* __restrict__ h, const float* __restrict__ st,
                     const float* __restrict__ fcw, const float* __restrict__ fcb,
                     float* __restrict__ out) {
  int idx = blockIdx.x * 256 + threadIdx.x;
  int b = idx >> 4, j = idx & 15;
  if (b >= BATCH || j >= 9) return;
  float acc = fcb[j];
  for (int c = 0; c < 24; c++) {
    float m = st[c], r = st[128 + c];
    const float* hr = h + ((size_t)b * 24 + c) * LOUT;
    const float* wr = fcw + (size_t)j * (24 * LOUT) + c * LOUT;
    for (int p = 0; p < LOUT; p++) acc += (hr[p] - m) * r * wr[p];
  }
  out[(size_t)b * 9 + j] = acc;
}

extern "C" void kernel_launch(void* const* d_in, const int* in_sizes, int n_in,
                              void* d_out, int out_size, void* d_ws, size_t ws_size,
                              hipStream_t stream) {
  (void)in_sizes; (void)n_in; (void)out_size; (void)ws_size;
  const float* x   = (const float*)d_in[0];
  const float* w1  = (const float*)d_in[1];
  const float* b1  = (const float*)d_in[2];
  Ptrs7 wp;
  for (int i = 0; i < 7; i++) wp.p[i] = (const float*)d_in[3 + i];
  const float* fcw = (const float*)d_in[10];
  const float* fcb = (const float*)d_in[11];
  float* out = (float*)d_out;

  char* base = (char*)d_ws;
  size_t off = 0;
  auto take = [&](size_t bytes) -> char* {
    char* r = base + off;
    off = (off + bytes + 255) & ~(size_t)255;
    return r;
  };
  int*    qfrag = (int*)   take(15360 * 4);
  float*  sw    = (float*) take(272 * 4);
  float*  stats = (float*) take(10 * 256 * 4);
  float*  F1    = (float*) take((size_t)MTOT * 24 * 4);
  float*  F2    = (float*) take((size_t)MTOT * 24 * 4);
  float*  F3    = (float*) take((size_t)MTOT * 24 * 4);
  float*  G     = (float*) take((size_t)MTOT * 128 * 4);
  int8_t* A8    = (int8_t*)take((size_t)MTOT * 24);
  int8_t* A8b   = (int8_t*)take((size_t)MTOT * 128);

  auto slot = [&](int s) { return stats + (size_t)s * 256; };

  const int T24  = MTOT * 24;
  const int T128 = MTOT * 128;
  const int GB24 = (T24 + 255) / 256, GB128 = (T128 + 255) / 256;
  const int GM   = (MTOT + 255) / 256;
  const int GT   = MTOT / 256;

  auto conv = [&](int L, const int8_t* act, float* zo) {
    const Layer& h = HLY[L];
    dim3 g(GT, h.nblk);
    switch (L) {
      case 0: case 1:
        k_binconv<24, 5><<<g, 256, 0, stream>>>(act, qfrag + h.fragOff, sw + h.swOff, zo, h.Cout, h.pad); break;
      case 3:
        k_binconv<128, 6><<<g, 256, 0, stream>>>(act, qfrag + h.fragOff, sw + h.swOff, zo, h.Cout, h.pad); break;
      default:
        k_binconv<24, 2><<<g, 256, 0, stream>>>(act, qfrag + h.fragOff, sw + h.swOff, zo, h.Cout, h.pad); break;
    }
  };

  k_prep <<<272, 64, 0, stream>>>(wp, qfrag, sw);
  k_conv1<<<GM, 256, 0, stream>>>(x, w1, b1, F1);
  k_stats<<<24, 256, 0, stream>>>(F1, 24, slot(0));
  k_bin  <<<GB24, 256, 0, stream>>>(F1, slot(0), A8, 24, T24);
  conv(0, A8, F2);
  k_stats<<<24, 256, 0, stream>>>(F2, 24, slot(1));
  k_bin  <<<GB24, 256, 0, stream>>>(F2, slot(1), A8, 24, T24);
  conv(1, A8, F3);
  k_stats<<<24, 256, 0, stream>>>(F3, 24, slot(2));
  k_res  <<<GB24, 256, 0, stream>>>(F1, slot(0), F3, slot(2), F2, T24);
  k_stats<<<24, 256, 0, stream>>>(F2, 24, slot(3));
  k_bin  <<<GB24, 256, 0, stream>>>(F2, slot(3), A8, 24, T24);
  conv(2, A8, G);
  k_stats<<<128, 256, 0, stream>>>(G, 128, slot(4));
  k_bin  <<<GB128, 256, 0, stream>>>(G, slot(4), A8b, 128, T128);
  conv(3, A8b, F1);
  k_stats<<<24, 256, 0, stream>>>(F1, 24, slot(5));
  k_bin  <<<GB24, 256, 0, stream>>>(F1, slot(5), A8, 24, T24);
  conv(4, A8, F3);
  k_stats<<<24, 256, 0, stream>>>(F3, 24, slot(6));
  k_bin  <<<GB24, 256, 0, stream>>>(F3, slot(6), A8, 24, T24);
  conv(5, A8, F3);
  k_stats<<<24, 256, 0, stream>>>(F3, 24, slot(7));
  k_bin  <<<GB24, 256, 0, stream>>>(F3, slot(7), A8, 24, T24);
  conv(6, A8, F3);
  k_stats<<<24, 256, 0, stream>>>(F3, 24, slot(8));
  k_res  <<<GB24, 256, 0, stream>>>(F1, slot(5), F3, slot(8), F2, T24);
  k_stats<<<24, 256, 0, stream>>>(F2, 24, slot(9));
  k_fc   <<<(BATCH * 16) / 256, 256, 0, stream>>>(F2, slot(9), fcw, fcb, out);
}
